// CrossLayerRouter_63067299775266
// MI455X (gfx1250) — matrix operand order repaired and checked by reading, not compiled, not run on hardware
//
#include <hip/hip_runtime.h>
#include <hip/hip_bf16.h>
#include <stdint.h>

// Problem constants (from the reference)
#define D_EMBED 2048
#define N_EXP   64
#define TOPK    8
#define TOKENS  32768              // B*S = 4*8192
#define R_ELEMS (TOKENS * N_EXP)   // router_output elements
#define I_ELEMS (TOKENS * TOPK)    // indices elements

typedef __attribute__((ext_vector_type(16))) __bf16 v16bf;
typedef __attribute__((ext_vector_type(8)))  float  v8f;
typedef __attribute__((ext_vector_type(4)))  float  v4f;
typedef __attribute__((ext_vector_type(8)))  unsigned int v8u;

union Tile16 {
  v16bf bf;
  v8u   u8;     // single 32B value -> contiguous 8-VGPR tuple
  __bf16 h[16];
};

#define WMMA_BF16(A, B, C) \
  __builtin_amdgcn_wmma_f32_16x16x32_bf16(false, (A), false, (B), (short)0, (C), false, false)

// ---------------------------------------------------------------------------
// Kernel 1: convert w_router / w_noise (f32, [2048 x 64]) into bf16 tiles in
// the exact CDNA5 B-matrix (32x16, KxN) VGPR layout, 1 KB per tile:
//   lane = 16*((K/8)%2) + N, vgpr r = 4*(K/16) + (K%8)/2, half = K%2
// Tile id t = mat*256 + kk*4 + nt
// ---------------------------------------------------------------------------
__global__ void moe_prep_weights(const float* __restrict__ wr,
                                 const float* __restrict__ wn,
                                 __bf16* __restrict__ wb) {
  const int t   = blockIdx.x;          // 0..511
  const int mat = t >> 8;
  const int kk  = (t & 255) >> 2;
  const int nt  = t & 3;
  const float* w = mat ? wn : wr;
  for (int e = threadIdx.x; e < 512; e += blockDim.x) {
    const int k = e >> 4;              // 0..31
    const int n = e & 15;              // 0..15
    const float v = w[(size_t)(kk * 32 + k) * N_EXP + nt * 16 + n];
    const int lane = (((k >> 3) & 1) << 4) + n;
    const int r    = ((k >> 4) << 2) + ((k & 7) >> 1);
    const int half = k & 1;
    wb[(size_t)t * 512 + lane * 16 + r * 2 + half] = (__bf16)v;
  }
}

// ---------------------------------------------------------------------------
// Kernel 2: fused router. 4 waves/block; each wave owns 32 tokens (two 16-row
// A tiles), software-pipelined B/x loads (ping-pong), 16 WMMAs per K-step.
// ---------------------------------------------------------------------------
__global__ __launch_bounds__(128) void moe_router(
    const float* __restrict__ X,     // [TOKENS, D]
    const float* __restrict__ EPS,   // [TOKENS, 64]
    const float* __restrict__ bR,    // [64]
    const float* __restrict__ bN,    // [64]
    const float* __restrict__ wS,    // [D]
    const float* __restrict__ bS,    // [1]
    const __bf16* __restrict__ WB,   // bf16 weight tiles (see prep)
    float* __restrict__ outR,        // [TOKENS, 64]
    int*   __restrict__ outI,        // [TOKENS, 8]
    float* __restrict__ outS)        // [TOKENS]
{
  __shared__ float lds[4 * 32 * N_EXP];   // 32 KB: per-wave 32x64 scratch

  const int lane = threadIdx.x & 31;
  const int wv   = threadIdx.x >> 5;
  const int hi   = lane >> 4;              // lane half (K sub-range select)
  const int nl   = lane & 15;              // N (for B/C) / M (for A) index
  const int tokenBase = blockIdx.x * 128 + wv * 32;

  // Per-lane byte bases (immediate offsets do the rest)
  const char* xb0 = (const char*)(X + (size_t)(tokenBase + nl)      * D_EMBED) + hi * 32;
  const char* xb1 = (const char*)(X + (size_t)(tokenBase + 16 + nl) * D_EMBED) + hi * 32;
  const char* wsb = (const char*)wS + hi * 32;
  const char* bb  = (const char*)WB + lane * 32;

  v8f accR0[4], accR1[4], accN0[4], accN1[4];
  #pragma unroll
  for (int i = 0; i < 4; ++i) {
    accR0[i] = (v8f){}; accR1[i] = (v8f){};
    accN0[i] = (v8f){}; accN1[i] = (v8f){};
  }
  v8f sacc0 = (v8f){}, sacc1 = (v8f){};    // vector skip-gate accumulators

  Tile16 b0[8], b1[8];
  v8f x0[4], x1[4], w0[2], w1[2];          // [row0:kb, row0:kb+16, row1:kb, row1:kb+16]

  auto loadB = [&](Tile16* b, int kk) {
    #pragma unroll
    for (int nt = 0; nt < 4; ++nt) {
      b[nt].u8     = *(const v8u*)(bb + (size_t)(kk * 4 + nt) * 1024);
      b[4 + nt].u8 = *(const v8u*)(bb + (size_t)(256 * 1024) +
                                   (size_t)(kk * 4 + nt) * 1024);
    }
  };
  auto loadX = [&](v8f* xv, v8f* wv, int kk) {
    const size_t o = (size_t)kk * 128;   // kk*32 floats
    xv[0] = __builtin_nontemporal_load((const v8f*)(xb0 + o));
    xv[1] = __builtin_nontemporal_load((const v8f*)(xb0 + o + 64));
    xv[2] = __builtin_nontemporal_load((const v8f*)(xb1 + o));
    xv[3] = __builtin_nontemporal_load((const v8f*)(xb1 + o + 64));
    wv[0] = *(const v8f*)(wsb + o);
    wv[1] = *(const v8f*)(wsb + o + 64);
  };
  auto compute = [&](const Tile16* b, const v8f* xv, const v8f* wv) {
    Tile16 a0, a1;
    #pragma unroll
    for (int i = 0; i < 8; ++i) {
      a0.h[i]     = (__bf16)xv[0][i];
      a0.h[8 + i] = (__bf16)xv[1][i];
      a1.h[i]     = (__bf16)xv[2][i];
      a1.h[8 + i] = (__bf16)xv[3][i];
    }
    // separate statements -> each contracts to a single v_pk_fma_f32
    sacc0 += xv[0] * wv[0];
    sacc0 += xv[1] * wv[1];
    sacc1 += xv[2] * wv[0];
    sacc1 += xv[3] * wv[1];
    // group WMMAs by shared A operand for back-to-back operand reuse
    #pragma unroll
    for (int t = 0; t < 4; ++t) accR0[t] = WMMA_BF16(a0.bf, b[t].bf,     accR0[t]);
    #pragma unroll
    for (int t = 0; t < 4; ++t) accN0[t] = WMMA_BF16(a0.bf, b[4 + t].bf, accN0[t]);
    #pragma unroll
    for (int t = 0; t < 4; ++t) accR1[t] = WMMA_BF16(a1.bf, b[t].bf,     accR1[t]);
    #pragma unroll
    for (int t = 0; t < 4; ++t) accN1[t] = WMMA_BF16(a1.bf, b[4 + t].bf, accN1[t]);
  };

  // Software pipeline: prefetch one K-step ahead into the other buffer set.
  loadB(b0, 0);
  loadX(x0, w0, 0);
  for (int kk = 0; kk < D_EMBED / 32; kk += 2) {
    loadB(b1, kk + 1);
    loadX(x1, w1, kk + 1);
    compute(b0, x0, w0);
    if (kk + 2 < D_EMBED / 32) {
      loadB(b0, kk + 2);
      loadX(x0, w0, kk + 2);
    }
    compute(b1, x1, w1);
  }

  // horizontal reduce of vector skip accumulators
  float skip0 = 0.f, skip1 = 0.f;
  #pragma unroll
  for (int i = 0; i < 8; ++i) { skip0 += sacc0[i]; skip1 += sacc1[i]; }
  // lane l and lane l^16 together cover all K of their rows
  skip0 += __shfl_xor(skip0, 16, 32);
  skip1 += __shfl_xor(skip1, 16, 32);
  const float skipDot = (lane < 16) ? skip0 : skip1;

  // ---- epilogue: bias + softplus + eps -> noisy logits into LDS ----
  // C layout: lane holds column N = nt*16 + nl; VGPR j holds row M = j + hi*8.
  float* wbuf = &lds[wv * 32 * N_EXP];
  #pragma unroll
  for (int g = 0; g < 2; ++g) {
    const v8f* aR = g ? accR1 : accR0;
    const v8f* aN = g ? accN1 : accN0;
    #pragma unroll
    for (int nt = 0; nt < 4; ++nt) {
      const int expert = nt * 16 + nl;
      const float rb = bR[expert];
      const float nb = bN[expert];
      #pragma unroll
      for (int j = 0; j < 8; ++j) {
        const int M = g * 16 + j + hi * 8;
        const int token = tokenBase + M;
        const float rl   = aR[nt][j] + rb;
        const float nlog = aN[nt][j] + nb;
        const float sp   = (nlog > 20.f) ? nlog : log1pf(__expf(nlog));
        const float noisy = rl + EPS[(size_t)token * N_EXP + expert] * sp;
        wbuf[M * N_EXP + expert] = noisy;
      }
    }
  }
  __syncthreads();

  // ---- top-8 + softmax: each lane owns one token ----
  float vals[TOPK], gate[TOPK];
  int   idxs[TOPK];
  {
    const float* rowp = &wbuf[lane * N_EXP];
    #pragma unroll
    for (int k = 0; k < TOPK; ++k) { vals[k] = -__builtin_inff(); idxs[k] = 0; }
    for (int e = 0; e < N_EXP; ++e) {
      const float v = rowp[e];
      if (v > vals[TOPK - 1]) {
        int p = TOPK - 1;
        #pragma unroll
        for (int j = TOPK - 2; j >= 0; --j)
          if (v > vals[j]) p = j;           // strict '>' keeps earlier index on ties
        #pragma unroll
        for (int j = TOPK - 1; j >= 1; --j)
          if (j > p) { vals[j] = vals[j - 1]; idxs[j] = idxs[j - 1]; }
        vals[p] = v; idxs[p] = e;
      }
    }
    const float mx = vals[0];
    float s = 0.f;
    #pragma unroll
    for (int k = 0; k < TOPK; ++k) { gate[k] = __expf(vals[k] - mx); s += gate[k]; }
    const float inv = 1.f / s;
    #pragma unroll
    for (int k = 0; k < TOPK; ++k) gate[k] *= inv;

    const int token = tokenBase + lane;
    #pragma unroll
    for (int k = 0; k < TOPK; ++k) outI[(size_t)token * TOPK + k] = idxs[k];
    outS[token] = 1.f / (1.f + __expf(-(skipDot + bS[0])));
  }
  __syncthreads();

  // ---- rebuild dense gate rows in LDS, then coalesced NT b128 stores ----
  for (int i = lane; i < 32 * N_EXP; i += 32) wbuf[i] = 0.f;
  __syncthreads();
  #pragma unroll
  for (int k = 0; k < TOPK; ++k) wbuf[lane * N_EXP + idxs[k]] = gate[k];
  __syncthreads();
  {
    const v4f* src = (const v4f*)wbuf;
    v4f* dst = (v4f*)(outR + (size_t)tokenBase * N_EXP);
    for (int i = lane; i < (32 * N_EXP) / 4; i += 32)
      __builtin_nontemporal_store(src[i], dst + i);
  }
}

// ---------------------------------------------------------------------------
extern "C" void kernel_launch(void* const* d_in, const int* in_sizes, int n_in,
                              void* d_out, int out_size, void* d_ws, size_t ws_size,
                              hipStream_t stream) {
  const float* x   = (const float*)d_in[0];
  const float* eps = (const float*)d_in[1];
  const float* wr  = (const float*)d_in[2];
  const float* br  = (const float*)d_in[3];
  const float* wn  = (const float*)d_in[4];
  const float* bn  = (const float*)d_in[5];
  const float* wsk = (const float*)d_in[6];
  const float* bsk = (const float*)d_in[7];

  float* out  = (float*)d_out;
  float* outR = out;                               // [TOKENS, 64]
  int*   outI = (int*)(out + R_ELEMS);             // [TOKENS, 8] int32
  float* outS = out + R_ELEMS + I_ELEMS;           // [TOKENS]

  __bf16* wb = (__bf16*)d_ws;                      // 512 KB bf16 weight tiles

  moe_prep_weights<<<512, 256, 0, stream>>>(wr, wn, wb);
  moe_router<<<TOKENS / 128, 128, 0, stream>>>(x, eps, br, bn, wsk, bsk, wb,
                                               outR, outI, outS);
}
